// SSN_29703993819824
// MI455X (gfx1250) — matrix operand order repaired and checked by reading, not compiled, not run on hardware
//
#include <hip/hip_runtime.h>
#include <hip/hip_bf16.h>

typedef __attribute__((ext_vector_type(16))) _Float16 v16h;
typedef __attribute__((ext_vector_type(8)))  _Float16 v8h;
typedef __attribute__((ext_vector_type(8)))  float    v8f;

typedef int v4i_ __attribute__((vector_size(16)));
typedef __attribute__((address_space(1))) v4i_ gv4i;   // global
typedef __attribute__((address_space(3))) v4i_ lv4i;   // LDS

#define HW (512 * 512)

// LDS layout (dynamic shared, in _Float16 units):
//   sA: input halo tile, [y:18][x:18][ci:136 (128 used)]  = 44064 halves
//   sB: 36 weight tiles, [tile*32+lane : pitch 24 halves] = 27648 halves
#define SA_HALVES   (18 * 18 * 136)
#define SB_HALVES   (36 * 32 * 24)
#define SMEM_BYTES  ((SA_HALVES + SB_HALVES) * 2)

#if __has_builtin(__builtin_amdgcn_global_load_async_to_lds_b128) && \
    __has_builtin(__builtin_amdgcn_s_wait_asynccnt)
#define USE_ASYNC_LDS 1
#endif

union V16U { v16h v; v8h h[2]; };

// ---------------------------------------------------------------------------
// Pre-swizzle conv weights into WMMA B-matrix tiles (f16), padded N=8 -> 16.
// 36 tiles = 9 taps x 4 K-chunks. Tile layout: [lane 0..31][e 0..15] f16,
// lane = {half = lane>>4, n = lane&15}, K = half*16 + e  (dense 16x16x32 B).
// ---------------------------------------------------------------------------
__global__ void prep_weights(const float* __restrict__ w, _Float16* __restrict__ btab) {
    int t = blockIdx.x * blockDim.x + threadIdx.x;
    if (t >= 36 * 32 * 16) return;
    int e    = t & 15;
    int lane = (t >> 4) & 31;
    int tile = t >> 9;
    int tap = tile >> 2, chunk = tile & 3;
    int ky = tap / 3, kx = tap % 3;
    int half = lane >> 4, n = lane & 15;
    int ci = chunk * 32 + half * 16 + e;
    float v = (n < 8) ? w[((n * 128 + ci) * 3 + ky) * 3 + kx] : 0.f;
    btab[t] = (_Float16)v;
}

// ---------------------------------------------------------------------------
// 3x3 conv (Cin=128 -> Cout=8) + bias + ReLU via v_wmma_f32_16x16x32_f16.
// Block = 16x16 output pixel tile, 8 waves x 2 rows. Input halo staged once
// into LDS as f16; A fragments become two ds_load_b128 per WMMA.
// ---------------------------------------------------------------------------
__global__ __launch_bounds__(256) void conv_wmma(
    const float* __restrict__ feat,      // (4,128,512,512)
    const float* __restrict__ bias,      // (8)
    const _Float16* __restrict__ btab,   // 36*32*16 f16 (B-fragment order)
    float* __restrict__ featall)         // (4,8,512,512)
{
    extern __shared__ _Float16 smem[];
    _Float16* sA = smem;
    _Float16* sB = smem + SA_HALVES;

    const int t    = threadIdx.x;
    const int lane = t & 31;
    const int wave = t >> 5;
    const int m    = lane & 15;
    const int half = lane >> 4;
    const int x0   = blockIdx.x * 16;
    const int y0   = blockIdx.y * 16;
    const int b    = blockIdx.z;

    // ---- Stage weight tiles into LDS (pitch 24 halves; data in first 16) ----
#if USE_ASYNC_LDS
    for (int p = t; p < 36 * 32 * 2; p += 256) {
        int tl = p >> 1, hi = p & 1;
        __builtin_amdgcn_global_load_async_to_lds_b128(
            (gv4i*)(btab + tl * 16 + hi * 8),
            (lv4i*)(sB + tl * 24 + hi * 8),
            0, 0);
    }
#else
    for (int p = t; p < 36 * 32 * 2; p += 256) {
        int tl = p >> 1, hi = p & 1;
        *(v8h*)(sB + tl * 24 + hi * 8) = *(const v8h*)(btab + tl * 16 + hi * 8);
    }
#endif

    // ---- Stage 18x18x128 input halo as f16 into LDS (zero-filled borders) ----
    const float* fb = feat + (size_t)b * 128 * HW;
    for (int l = t; l < 128 * 18 * 18; l += 256) {
        int x  = l % 18;
        int r  = l / 18;
        int yy = r % 18;
        int ci = r / 18;
        int gy = y0 - 1 + yy;
        int gx = x0 - 1 + x;
        float v = ((unsigned)gy < 512u && (unsigned)gx < 512u)
                      ? fb[(size_t)ci * HW + (size_t)gy * 512 + gx] : 0.f;
        sA[(yy * 18 + x) * 136 + ci] = (_Float16)v;
    }
#if USE_ASYNC_LDS
    __builtin_amdgcn_s_wait_asynccnt(0);
#endif
    __syncthreads();

    // ---- Main WMMA loop: 9 taps x 4 K-chunks x 2 rows = 72 WMMAs / wave ----
    v8f acc0 = {}, acc1 = {};
    const int yr = wave * 2;

    for (int tap = 0; tap < 9; ++tap) {
        int ky = tap / 3, kx = tap % 3;
        for (int chunk = 0; chunk < 4; ++chunk) {
            int c0 = chunk * 32;

            const _Float16* bp = sB + (size_t)((tap * 4 + chunk) * 32 + lane) * 24;
            V16U bt; bt.h[0] = *(const v8h*)bp; bt.h[1] = *(const v8h*)(bp + 8);

            // A fragment: [y][x][ci] layout gives two contiguous 8-halve runs
            const _Float16* ap0 = sA + ((size_t)((yr + ky) * 18 + (m + kx)) * 136
                                        + c0 + half * 16);
            V16U a0; a0.h[0] = *(const v8h*)ap0; a0.h[1] = *(const v8h*)(ap0 + 8);
            acc0 = __builtin_amdgcn_wmma_f32_16x16x32_f16(
                false, a0.v, false, bt.v, (short)0, acc0, false, false);

            const _Float16* ap1 = ap0 + 18 * 136;   // next y row
            V16U a1; a1.h[0] = *(const v8h*)ap1; a1.h[1] = *(const v8h*)(ap1 + 8);
            acc1 = __builtin_amdgcn_wmma_f32_16x16x32_f16(
                false, a1.v, false, bt.v, (short)0, acc1, false, false);
        }
    }

    // ---- Epilogue: C/D layout -> VGPR r holds M = r + 8*half, N = lane&15 ----
    int co = lane & 15;
    if (co < 8) {
        float bv = bias[co];
        float* obase = featall + ((size_t)b * 8 + co) * HW;
        float* orow0 = obase + (size_t)(y0 + yr) * 512;
        float* orow1 = orow0 + 512;
#pragma unroll
        for (int r = 0; r < 8; ++r) {
            int xx = x0 + r + half * 8;
            float v0 = acc0[r] + bv;
            float v1 = acc1[r] + bv;
            orow0[xx] = v0 > 0.f ? v0 : 0.f;
            orow1[xx] = v1 > 0.f ? v1 : 0.f;
        }
    }
}

// ---------------------------------------------------------------------------
// Initial superpixel means: each superpixel is exactly a 16x16 pixel block.
// ---------------------------------------------------------------------------
__global__ __launch_bounds__(256) void spf_init(
    const float* __restrict__ featall, float* __restrict__ spf)
{
    __shared__ float acc[8];
    int t = threadIdx.x;
    if (t < 8) acc[t] = 0.f;
    __syncthreads();
    int b = blockIdx.z, qy = blockIdx.y, qx = blockIdx.x;
    int y = qy * 16 + (t >> 4), x = qx * 16 + (t & 15);
    const float* fp = featall + (size_t)b * 8 * HW + (size_t)y * 512 + x;
#pragma unroll
    for (int c = 0; c < 8; ++c)
        atomicAdd(&acc[c], fp[(size_t)c * HW]);
    __syncthreads();
    if (t < 8)
        spf[(b * 8 + t) * 1024 + qy * 32 + qx] = acc[t] * (1.f / 256.f);
}

// ---------------------------------------------------------------------------
// Per-pixel 9-neighbor distance + masked softmax -> Q (4,9,512,512).
// ---------------------------------------------------------------------------
__global__ __launch_bounds__(256) void q_kernel(
    const float* __restrict__ featall, const float* __restrict__ spf,
    float* __restrict__ Q)
{
    __shared__ float sn[9][8];
    __shared__ int   svalid[9];
    int t = threadIdx.x;
    int b = blockIdx.z, qy = blockIdx.y, qx = blockIdx.x;
    if (t < 72) {
        int j = t >> 3, c = t & 7;
        int ny = qy + (j / 3 - 1), nx = qx + (j % 3 - 1);
        bool ok = (ny >= 0) && (ny < 32) && (nx >= 0) && (nx < 32);
        sn[j][c] = ok ? spf[(b * 8 + c) * 1024 + ny * 32 + nx] : 0.f;
        if (c == 0) svalid[j] = ok ? 1 : 0;
    }
    __syncthreads();

    int y = qy * 16 + (t >> 4), x = qx * 16 + (t & 15);
    const float* fp = featall + (size_t)b * 8 * HW + (size_t)y * 512 + x;
    float f[8];
#pragma unroll
    for (int c = 0; c < 8; ++c) f[c] = fp[(size_t)c * HW];

    float nd[9], mx = -1e30f;
#pragma unroll
    for (int j = 0; j < 9; ++j) {
        float s = 0.f;
#pragma unroll
        for (int c = 0; c < 8; ++c) { float d = f[c] - sn[j][c]; s += d * d; }
        nd[j] = svalid[j] ? -s : -1e30f;
        mx = fmaxf(mx, nd[j]);
    }
    float e[9], sum = 0.f;
#pragma unroll
    for (int j = 0; j < 9; ++j) {
        e[j] = svalid[j] ? __expf(nd[j] - mx) : 0.f;  // invalid -> exact 0
        sum += e[j];
    }
    float inv = 1.f / sum;
    float* qp = Q + (size_t)b * 9 * HW + (size_t)y * 512 + x;
#pragma unroll
    for (int j = 0; j < 9; ++j) qp[(size_t)j * HW] = e[j] * inv;
}

// ---------------------------------------------------------------------------
// spf update as a GATHER (no global atomics).
// ---------------------------------------------------------------------------
__global__ __launch_bounds__(256) void update_kernel(
    const float* __restrict__ featall, const float* __restrict__ Q,
    float* __restrict__ spf)
{
    __shared__ float acc[9];   // [0..7]=num, [8]=den
    int t = threadIdx.x;
    if (t < 9) acc[t] = 0.f;
    __syncthreads();
    int b = blockIdx.z, qy = blockIdx.y, qx = blockIdx.x;

    float pn[8] = {0.f, 0.f, 0.f, 0.f, 0.f, 0.f, 0.f, 0.f};
    float pd = 0.f;
#pragma unroll
    for (int dy = -1; dy <= 1; ++dy)
#pragma unroll
        for (int dx = -1; dx <= 1; ++dx) {
            int py = qy - dy, px = qx - dx;
            if (py < 0 || py >= 32 || px < 0 || px >= 32) continue;
            int j = (dy + 1) * 3 + (dx + 1);
            int y = py * 16 + (t >> 4), x = px * 16 + (t & 15);
            size_t poff = (size_t)y * 512 + x;
            float q = Q[((size_t)b * 9 + j) * HW + poff];
            pd += q;
            const float* fp = featall + (size_t)b * 8 * HW + poff;
#pragma unroll
            for (int c = 0; c < 8; ++c) pn[c] += q * fp[(size_t)c * HW];
        }
#pragma unroll
    for (int c = 0; c < 8; ++c) atomicAdd(&acc[c], pn[c]);
    atomicAdd(&acc[8], pd);
    __syncthreads();
    if (t < 8) {
        float den = fmaxf(acc[8], 1e-12f);
        spf[(b * 8 + t) * 1024 + qy * 32 + qx] = acc[t] / den;
    }
}

// ---------------------------------------------------------------------------
extern "C" void kernel_launch(void* const* d_in, const int* in_sizes, int n_in,
                              void* d_out, int out_size, void* d_ws, size_t ws_size,
                              hipStream_t stream) {
    const float* feat = (const float*)d_in[0];   // (4,128,512,512)
    const float* w    = (const float*)d_in[1];   // (8,128,3,3)
    const float* bias = (const float*)d_in[2];   // (8)

    float* out     = (float*)d_out;
    float* Q       = out;                                  // (4,9,512,512)
    float* spf     = out + (size_t)4 * 9 * HW;             // (4,8,1024)
    float* featall = spf + (size_t)4 * 8 * 1024;           // (4,8,512,512)

    _Float16* btab = (_Float16*)d_ws;                      // 36 KB weight tiles

    prep_weights<<<(36 * 32 * 16 + 255) / 256, 256, 0, stream>>>(w, btab);
    conv_wmma<<<dim3(32, 32, 4), 256, SMEM_BYTES, stream>>>(feat, bias, btab, featall);
    spf_init<<<dim3(32, 32, 4), 256, 0, stream>>>(featall, spf);
    for (int it = 0; it < 5; ++it) {
        q_kernel<<<dim3(32, 32, 4), 256, 0, stream>>>(featall, spf, Q);
        update_kernel<<<dim3(32, 32, 4), 256, 0, stream>>>(featall, Q, spf);
    }
}
